// SparseAttention_20538533610133
// MI455X (gfx1250) — matrix operand order repaired and checked by reading, not compiled, not run on hardware
//
#include <hip/hip_runtime.h>

typedef __attribute__((ext_vector_type(4)))  float    v4f;
typedef __attribute__((ext_vector_type(8)))  float    v8f;
typedef __attribute__((ext_vector_type(4)))  _Float16 v4h;
typedef __attribute__((ext_vector_type(8)))  _Float16 v8h;
typedef __attribute__((ext_vector_type(16))) _Float16 v16h;

#define SEQ     128
#define DMODEL  1024
#define NHEADS  16
#define HDIM    64
#define BAND    4
#define SCALE   0.125f      // HEAD_DIM^-0.5
#define LSTR    72          // LDS row stride in halves (16B-aligned rows, conflict-free banks)

// LDS offsets in halves
#define QS_OFF  0
#define KS_OFF  (16 * LSTR)             // Q tile: 16 x 64
#define VT_OFF  (KS_OFF + 48 * LSTR)    // K window: 48 x 64
#define PS_OFF  (VT_OFF + 64 * LSTR)    // V^T: 64 dims x 64 keys (padded window)
#define LDS_HALVES (PS_OFF + 16 * LSTR) // P: 16 x 64 (cols 48..63 zero)

// Build a v16h A/B fragment from two 16-byte LDS reads.
static __device__ __forceinline__ v16h ld_ab(const _Float16* p0, const _Float16* p1) {
  v8h a = *(const v8h*)p0;
  v8h b = *(const v8h*)p1;
  v16h r;
#pragma unroll
  for (int i = 0; i < 8; ++i) { r[i] = a[i]; r[i + 8] = b[i]; }
  return r;
}

__global__ __launch_bounds__(32) void band_attn_wmma(
    const float* __restrict__ Q, const float* __restrict__ K,
    const float* __restrict__ V, float* __restrict__ O) {
  __shared__ _Float16 lds[LDS_HALVES];
  _Float16* qs = lds + QS_OFF;
  _Float16* ks = lds + KS_OFF;
  _Float16* vt = lds + VT_OFF;
  _Float16* ps = lds + PS_OFF;

  const int lane = threadIdx.x;
  const int ln   = lane & 15;   // N / row-in-tile index
  const int hf   = lane >> 4;   // which 16-lane half

  const int qt = blockIdx.x & 7;
  const int hd = (blockIdx.x >> 3) & (NHEADS - 1);
  const int b  = blockIdx.x >> 7;

  const int qbase = qt * 16;
  const int kbase = qbase - 16;   // key-window start (may be <0, clamped on load)

  const size_t base = (size_t)b * SEQ * DMODEL + (size_t)hd * HDIM;
  const float* qg = Q + base;
  const float* kg = K + base;
  const float* vg = V + base;
  float*       og = O + base;

  const int c4 = ln * 4;          // 4-float column chunk this lane stages

  // ---- stage Q tile (pre-scaled), coalesced: one b128/lane covers 2 rows ----
#pragma unroll
  for (int it = 0; it < 8; ++it) {
    int r = it * 2 + hf;
    v4f x = *(const v4f*)(qg + (size_t)(qbase + r) * DMODEL + c4);
    x *= SCALE;
    *(v4h*)(qs + r * LSTR + c4) = __builtin_convertvector(x, v4h);
  }
  // ---- stage K window (48 rows, row-clamped; masked later) ----
#pragma unroll
  for (int it = 0; it < 24; ++it) {
    int w = it * 2 + hf;
    int j = kbase + w; j = j < 0 ? 0 : (j > SEQ - 1 ? SEQ - 1 : j);
    v4f x = *(const v4f*)(kg + (size_t)j * DMODEL + c4);
    *(v4h*)(ks + w * LSTR + c4) = __builtin_convertvector(x, v4h);
  }
  // ---- stage V^T (64-key padded window, transposed during store) ----
#pragma unroll
  for (int it = 0; it < 32; ++it) {
    int w = it * 2 + hf;
    int j = kbase + w; j = j < 0 ? 0 : (j > SEQ - 1 ? SEQ - 1 : j);
    v4f x = *(const v4f*)(vg + (size_t)j * DMODEL + c4);
#pragma unroll
    for (int t = 0; t < 4; ++t) vt[(c4 + t) * LSTR + w] = (_Float16)x[t];
  }
  __syncthreads();

  // ---- S = (Q*scale) @ K^T over 3 key tiles: 2 WMMAs (K=32 each) per tile ----
  v8f s[3];
#pragma unroll
  for (int kt = 0; kt < 3; ++kt) {
    v8f acc = {};
#pragma unroll
    for (int ch = 0; ch < 2; ++ch) {
      const int db = ch * 32;
      v16h a = ld_ab(qs + ln * LSTR + db + 8 * hf,
                     qs + ln * LSTR + db + 16 + 8 * hf);
      const _Float16* bp = ks + (kt * 16 + ln) * LSTR + db + 8 * hf;
      v16h bb = ld_ab(bp, bp + 16);
      acc = __builtin_amdgcn_wmma_f32_16x16x32_f16(false, a, false, bb,
                                                   (short)0, acc, false, false);
    }
    s[kt] = acc;
  }

  // ---- band mask + softmax (row reductions via 16-lane shfl_xor) ----
  float rinv[8];
#pragma unroll
  for (int r = 0; r < 8; ++r) {
    const int i = qbase + r + hf * 8;     // C/D layout: VGPR r holds rows r and r+8
    float e0, e1, e2;
    {
      int j0 = kbase + ln, j1 = j0 + 16, j2 = j0 + 32;
      e0 = (j0 >= 0 && j0 < SEQ && (unsigned)(i - j0 + BAND) <= 2u * BAND) ? s[0][r] : -__builtin_inff();
      e1 = (j1 >= 0 && j1 < SEQ && (unsigned)(i - j1 + BAND) <= 2u * BAND) ? s[1][r] : -__builtin_inff();
      e2 = (j2 >= 0 && j2 < SEQ && (unsigned)(i - j2 + BAND) <= 2u * BAND) ? s[2][r] : -__builtin_inff();
    }
    float m = fmaxf(fmaxf(e0, e1), e2);
#pragma unroll
    for (int msk = 8; msk >= 1; msk >>= 1) m = fmaxf(m, __shfl_xor(m, msk, 32));
    e0 = __expf(e0 - m); e1 = __expf(e1 - m); e2 = __expf(e2 - m);
    float sum = e0 + e1 + e2;
#pragma unroll
    for (int msk = 8; msk >= 1; msk >>= 1) sum += __shfl_xor(sum, msk, 32);
    rinv[r] = 1.0f / sum;                 // diagonal always in band -> sum > 0
    const int prow = (r + hf * 8) * LSTR; // re-layout P: C/D layout -> row-major LDS
    ps[prow +  0 + ln] = (_Float16)e0;
    ps[prow + 16 + ln] = (_Float16)e1;
    ps[prow + 32 + ln] = (_Float16)e2;
    ps[prow + 48 + ln] = (_Float16)0.f;   // zero pad cols 48..63 (kills clamped V rows)
  }
  __syncthreads();

  // ---- O = P @ V over padded K=64: 2 WMMAs per 16-dim output tile ----
#pragma unroll
  for (int nt = 0; nt < 4; ++nt) {
    v8f o = {};
#pragma unroll
    for (int ch = 0; ch < 2; ++ch) {
      const int cb = ch * 32;
      v16h a = ld_ab(ps + ln * LSTR + cb + 8 * hf,
                     ps + ln * LSTR + cb + 16 + 8 * hf);
      const _Float16* bp = vt + (nt * 16 + ln) * LSTR + cb + 8 * hf;
      v16h bb = ld_ab(bp, bp + 16);
      o = __builtin_amdgcn_wmma_f32_16x16x32_f16(false, a, false, bb,
                                                 (short)0, o, false, false);
    }
#pragma unroll
    for (int r = 0; r < 8; ++r)
      og[(size_t)(qbase + r + hf * 8) * DMODEL + nt * 16 + ln] = o[r] * rinv[r];
  }
}

extern "C" void kernel_launch(void* const* d_in, const int* in_sizes, int n_in,
                              void* d_out, int out_size, void* d_ws, size_t ws_size,
                              hipStream_t stream) {
  const float* q = (const float*)d_in[0];
  const float* k = (const float*)d_in[1];
  const float* v = (const float*)d_in[2];
  float* o = (float*)d_out;
  const int B = in_sizes[0] / (SEQ * DMODEL);          // batch from flat size
  dim3 grid((unsigned)(B * NHEADS * (SEQ / 16)));       // one wave per (b, h, 16-query tile)
  band_attn_wmma<<<grid, dim3(32), 0, stream>>>(q, k, v, o);
  (void)n_in; (void)out_size; (void)d_ws; (void)ws_size;
}
